// MACEModel_72335839199641
// MI455X (gfx1250) — matrix operand order repaired and checked by reading, not compiled, not run on hardware
//
#include <hip/hip_runtime.h>
#include <hip/hip_bf16.h>
#include <math.h>

// ---------------- problem constants ----------------
#define BB 2
#define NN 128
#define EE 2048
#define CC 64
#define BE (BB*EE)
#define TCOLS 35   // total t columns over the 11 paths (layer0 uses first 9)
#define ET 32      // edges per workgroup in the WMMA conv kernel

// PATHS1 metadata (layer0 = first 3 entries)
constexpr int PL1c[11]  = {0,0,0,1,1,1,1,2,2,2,2};
constexpr int PL2c[11]  = {0,1,2,0,1,1,2,0,1,2,2};
constexpr int PL3c[11]  = {0,1,2,1,0,2,1,2,1,0,2};
constexpr int TCOL0c[11]= {0,1,4,9,12,13,18,21,26,29,30};
constexpr int DIMSc[3]  = {1,3,5};
constexpr int OFFc[3]   = {0,1,4};
// PATHS3: pairs (p1,p2) with PATHS1[p1].l3 == PATHS1[p2].l1, reference order
constexpr int P3Ac[41] = {0,0,0, 1,1,1,1, 2,2,2,2, 3,3,3,3, 4,4,4, 5,5,5,5,
                          6,6,6,6, 7,7,7,7, 8,8,8,8, 9,9,9, 10,10,10,10};
constexpr int P3Bc[41] = {0,1,2, 3,4,5,6, 7,8,9,10, 3,4,5,6, 0,1,2, 7,8,9,10,
                          3,4,5,6, 7,8,9,10, 3,4,5,6, 0,1,2, 7,8,9,10};

typedef __bf16 bf16x16 __attribute__((ext_vector_type(16)));
typedef float  floatx8 __attribute__((ext_vector_type(8)));

union Frag16 { bf16x16 bf; uint4 q[2]; unsigned short us[16]; };

template<int V> struct IC { static constexpr int value = V; };

__device__ __forceinline__ unsigned short f2bf(float f) {
    unsigned int u = __float_as_uint(f);
    unsigned int r = (u + 0x7FFFu + ((u >> 16) & 1u)) >> 16;   // RNE
    return (unsigned short)r;
}

// ---------------- CG (real Wigner 3j) computed on device ----------------
__device__ double dfact(int n){ double r=1.0; for(int i=2;i<=n;i++) r*=(double)i; return r; }

__device__ double w3j_elem(int l1,int l2,int l3,int m1,int m2,int m3){
    if (m1+m2+m3 != 0) return 0.0;
    int e = l1 - l2 - m3;
    double sgn = (((e % 2) + 2) % 2) ? -1.0 : 1.0;
    double pref = sgn * sqrt(dfact(l1+l2-l3)*dfact(l1-l2+l3)*dfact(-l1+l2+l3)/dfact(l1+l2+l3+1)
                 * dfact(l1+m1)*dfact(l1-m1)*dfact(l2+m2)*dfact(l2-m2)*dfact(l3+m3)*dfact(l3-m3));
    int t0 = 0; if (l2-l3-m1 > t0) t0 = l2-l3-m1; if (l1-l3+m2 > t0) t0 = l1-l3+m2;
    int t1 = l1+l2-l3; if (l1-m1 < t1) t1 = l1-m1; if (l2+m2 < t1) t1 = l2+m2;
    double s = 0.0;
    for (int t=t0; t<=t1; t++){
        double d = dfact(t)*dfact(l3-l2+t+m1)*dfact(l3-l1+t-m2)*dfact(l1+l2-l3-t)
                 * dfact(l1-t-m1)*dfact(l2-t+m2);
        s += ((t & 1) ? -1.0 : 1.0) / d;
    }
    return pref * s;
}

__device__ void qelem(int l,int row,int col,double* pre,double* pim){
    const double s = 0.70710678118654752440;
    double r=0.0, i=0.0;
    int m = row - l;
    if (m < 0) { if (col == l - m) r =  s; else if (col == l + m) i = -s; }
    else if (m == 0) { if (col == l) r = 1.0; }
    else { double g = (m & 1) ? -1.0 : 1.0;
           if (col == l + m) r = g*s; else if (col == l - m) i = g*s; }
    // multiply by (-1j)^l
    int lm = l & 3; double rr, ii;
    if (lm==0){rr=r; ii=i;} else if (lm==1){rr=i; ii=-r;}
    else if (lm==2){rr=-r; ii=-i;} else {rr=-i; ii=r;}
    *pre = rr; *pim = ii;
}

__global__ void k_cg(float* __restrict__ cg){
    int p = blockIdx.x;
    int t = threadIdx.x;
    if (t >= 125) return;
    int l1 = PL1c[p], l2 = PL2c[p], l3 = PL3c[p];
    int d1 = 2*l1+1, d2 = 2*l2+1, d3 = 2*l3+1;
    int i = t/25, j = (t/5)%5, k = t%5;
    float out = 0.f;
    if (i < d1 && j < d2 && k < d3){
        double acc = 0.0;
        for (int a=0; a<d1; a++)
        for (int b=0; b<d2; b++)
        for (int n=0; n<d3; n++){
            double w = w3j_elem(l1,l2,l3, a-l1, b-l2, n-l3);
            if (w == 0.0) continue;
            double r1,i1,r2,i2,r3,i3;
            qelem(l1,i,a,&r1,&i1);
            qelem(l2,j,b,&r2,&i2);
            qelem(l3,k,n,&r3,&i3);
            double rr = r1*r2 - i1*i2;
            double ri = r1*i2 + i1*r2;
            double realpart = rr*r3 - ri*i3;
            acc += realpart * w;
        }
        out = (float)acc;
    }
    cg[p*125 + t] = out;
}

// ---------------- geometry: Y (l<=2 real SH) + Bessel radial ----------------
__global__ void k_geom(const float* __restrict__ pos, const int* __restrict__ edges,
                       float* __restrict__ Y, float* __restrict__ ef){
    int idx = blockIdx.x*blockDim.x + threadIdx.x;
    if (idx >= BE) return;
    int b = idx / EE, e = idx % EE;
    int s = edges[2*e + 0], d = edges[2*e + 1];
    float vx = pos[(b*NN+s)*3+0] - pos[(b*NN+d)*3+0];
    float vy = pos[(b*NN+s)*3+1] - pos[(b*NN+d)*3+1];
    float vz = pos[(b*NN+s)*3+2] - pos[(b*NN+d)*3+2];
    float r = sqrtf(vx*vx + vy*vy + vz*vz);
    r = fmaxf(r, 1e-9f);
    float x = vx/r, y = vy/r, z = vz/r;
    const float s3 = 1.7320508075688772f, s15 = 3.8729833462074170f, s5 = 2.2360679774997896f;
    float* Yp = Y + (size_t)idx*9;
    Yp[0] = 1.f;       Yp[1] = s3*y;      Yp[2] = s3*z;
    Yp[3] = s3*x;      Yp[4] = s15*x*y;   Yp[5] = s15*y*z;
    Yp[6] = 0.5f*s5*(3.f*z*z - 1.f);
    Yp[7] = s15*z*x;   Yp[8] = 0.5f*s15*(x*x - y*y);
    float u = r / 5.0f;
    float u5 = u*u*u*u*u;
    float env = 1.f - 21.f*u5 + 35.f*u5*u - 15.f*u5*u*u;
    float pre = sqrtf(0.4f)/r * ((u < 1.f) ? env : 0.f);
    const float PIov5 = 0.62831853071795864769f;
    for (int kk=1; kk<=8; kk++)
        ef[(size_t)idx*8 + kk-1] = pre * sinf((float)kk * PIov5 * r);
}

// ---------------- misc small kernels ----------------
__global__ void k_init_h(const int* __restrict__ at, const float* __restrict__ emb,
                         float* __restrict__ h){
    int idx = blockIdx.x*blockDim.x + threadIdx.x;
    if (idx >= BB*NN*CC) return;
    int c = idx % CC, bn = idx / CC;
    float v = emb[at[bn]*CC + c];
    float* hp = h + (size_t)idx*9;
    hp[0] = v;
    #pragma unroll
    for (int i=1;i<9;i++) hp[i] = 0.f;
}

__global__ void k_zero(float* __restrict__ p, int n){
    int i = blockIdx.x*blockDim.x + threadIdx.x;
    if (i < n) p[i] = 0.f;
}

__global__ void k_hid(const float* __restrict__ ef, const float* __restrict__ w,
                      const float* __restrict__ bvec, unsigned short* __restrict__ hid){
    int idx = blockIdx.x*blockDim.x + threadIdx.x;
    if (idx >= BE*CC) return;
    int be = idx / CC, c = idx % CC;
    float s = bvec[c];
    #pragma unroll
    for (int k=0;k<8;k++) s += ef[(size_t)be*8 + k] * w[k*CC + c];
    hid[idx] = f2bf(fmaxf(s, 0.f));
}

// fc2_w (64 x numel, f32) -> bf16 in per-lane WMMA B-fragment tile layout
__global__ void k_wconv(const float* __restrict__ w, unsigned short* __restrict__ wf, int numel){
    int total = numel * CC;
    for (int f = blockIdx.x*blockDim.x + threadIdx.x; f < total; f += gridDim.x*blockDim.x){
        int tile = f >> 10;            // 1024 bf16 per 16-wide N tile
        int r    = f & 1023;
        int half = r >> 9;             // K-step (c 0..31 / 32..63)
        int r2   = r & 511;
        int L    = r2 >> 4;            // lane
        int j    = r2 & 15;            // fragment element
        int c    = ((L < 16) ? 0 : 8) + 32*half + ((j < 8) ? j : 8 + j);
        int n    = tile*16 + (L & 15);
        wf[f] = f2bf(w[(size_t)c*numel + n]);
    }
}

// per-edge CG tensor product: t[be,u,col] = sum hs[u,i1]*Y[i2]*CG
__global__ void k_t(const float* __restrict__ h, const float* __restrict__ Y,
                    const int* __restrict__ edges, const float* __restrict__ cg,
                    float* __restrict__ t, int npaths){
    __shared__ float cgS[11*125];
    for (int x = threadIdx.x; x < 11*125; x += blockDim.x) cgS[x] = cg[x];
    __syncthreads();
    int idx = blockIdx.x*blockDim.x + threadIdx.x;
    if (idx >= BE*CC) return;
    int be = idx / CC, u = idx % CC;
    int b = be / EE, e = be % EE;
    int s = edges[2*e];
    const float* hp = h + ((size_t)(b*NN + s)*CC + u)*9;
    float hs[9], Yv[9];
    #pragma unroll
    for (int i=0;i<9;i++){ hs[i] = hp[i]; Yv[i] = Y[(size_t)be*9 + i]; }
    float* tp = t + (size_t)idx*TCOLS;
    for (int p=0; p<npaths; p++){
        int l1 = PL1c[p], l2 = PL2c[p], l3 = PL3c[p];
        int d1 = DIMSc[l1], d2 = DIMSc[l2], d3 = DIMSc[l3];
        int o1 = OFFc[l1], o2 = OFFc[l2];
        const float* cgp = &cgS[p*125];
        for (int k=0;k<d3;k++){
            float acc = 0.f;
            for (int i=0;i<d1;i++)
                for (int j=0;j<d2;j++)
                    acc += hs[o1+i]*Yv[o2+j]*cgp[(i*5+j)*5 + k];
            tp[TCOL0c[p] + k] = acc;
        }
    }
}

// ---------------- fused WMMA conv: wp = hid@W + b, applied to t, scattered ----------------
// 32 edges per block, 8 waves: wave w owns (k-tile = w&3, M-tile = w>>2).
// The two waves sharing a k-tile fetch identical B fragments (absorbed by the
// WGP vector cache), so per-WGP L2 weight traffic is halved vs 16-edge tiles.
__global__ void __launch_bounds__(256) k_conv(
    const unsigned short* __restrict__ Wfrag,
    const float* __restrict__ fc2b,
    const unsigned short* __restrict__ hid,
    const float* __restrict__ tmat,
    const int* __restrict__ edges,
    float* __restrict__ msg,
    int npaths, float n0, float n1, float n2)
{
    __shared__ unsigned short hidS[ET*CC];   // 4 KB  (A tile, bf16)
    __shared__ float tS[ET*CC*5];            // 40 KB (current path t tile)

    const int tid   = threadIdx.x;
    const int tileE = blockIdx.x;            // BE/ET tiles of ET edges
    const int b     = tileE / (EE/ET);
    const int e0    = (tileE % (EE/ET)) * ET;
    const size_t be0 = (size_t)b*EE + e0;

    for (int x = tid; x < ET*CC; x += 256) hidS[x] = hid[be0*CC + x];
    __syncthreads();

    const int lane  = tid & 31;
    const int wv    = tid >> 5;              // 8 waves
    const int kt    = wv & 3;                // k-tile
    const int mt    = wv >> 2;               // M-tile (edge half)
    const int m     = lane & 15;
    const int hi    = (lane < 16) ? 0 : 1;
    const int mhalf = hi * 8;
    const int erow0 = mt*16 + mhalf;         // first edge row this lane covers

    // A fragments (hid rows, bf16, 16x32 per K-step) from LDS
    Frag16 a0, a1;
    {
        const uint4* hrow = reinterpret_cast<const uint4*>(&hidS[(mt*16 + m)*CC]);
        a0.q[0] = hrow[0 + hi];
        a0.q[1] = hrow[2 + hi];
        a1.q[0] = hrow[4 + hi];
        a1.q[1] = hrow[6 + hi];
    }

    float accR[8][9];
    #pragma unroll
    for (int v=0;v<8;v++)
        #pragma unroll
        for (int i=0;i<9;i++) accR[v][i] = 0.f;

    auto run_path = [&](int p, auto dimc, auto ioffc){
        constexpr int DIM  = decltype(dimc)::value;
        constexpr int IOFF = decltype(ioffc)::value;
        const uint4* base = reinterpret_cast<const uint4*>(Wfrag);
        const int nb = p*256 + kt;           // tile index base (n-tiles of 16)
        for (int u=0; u<CC; ++u){
            const uint4* bp = base + (size_t)(nb + u*4) * 128;   // 2 KB per tile
            __builtin_prefetch(bp + 4*128, 0, 0);                // next u
            Frag16 b0f, b1f;
            b0f.q[0] = bp[lane*2 + 0];
            b0f.q[1] = bp[lane*2 + 1];
            b1f.q[0] = bp[64 + lane*2 + 0];
            b1f.q[1] = bp[64 + lane*2 + 1];
            float bias = fc2b[p*4096 + u*64 + kt*16 + m];
            floatx8 cf;
            #pragma unroll
            for (int z=0;z<8;z++) cf[z] = bias;
            cf = __builtin_amdgcn_wmma_f32_16x16x32_bf16(false, a0.bf, false, b0f.bf,
                                                         (short)0, cf, false, false);
            cf = __builtin_amdgcn_wmma_f32_16x16x32_bf16(false, a1.bf, false, b1f.bf,
                                                         (short)0, cf, false, false);
            #pragma unroll
            for (int v=0;v<8;v++){
                const float wvv = cf[v];                   // wp[e=erow0+v, u, k]
                const float* trow = &tS[((erow0 + v)*CC + u)*DIM];
                #pragma unroll
                for (int i=0;i<DIM;i++) accR[v][IOFF+i] += wvv * trow[i];
            }
        }
    };

    for (int p=0; p<npaths; p++){
        const int l3  = PL3c[p];
        const int dim = DIMSc[l3];
        const int tc  = TCOL0c[p];
        __syncthreads();                      // previous path done reading tS
        const int tot = ET*CC*dim;
        for (int x = tid; x < tot; x += 256){
            const int e   = x / (CC*dim);
            const int rem = x - e*(CC*dim);
            const int u   = rem / dim;
            const int i   = rem - u*dim;
            tS[x] = tmat[((be0 + e)*CC + u)*TCOLS + tc + i];
        }
        __syncthreads();
        if (l3 == 0)      run_path(p, IC<1>{}, IC<0>{});
        else if (l3 == 1) run_path(p, IC<3>{}, IC<1>{});
        else              run_path(p, IC<5>{}, IC<4>{});
    }

    // segment-sum into msg[b, dst, k, i] with per-l normalization
    const int k = kt*16 + m;
    const float nrm[9] = {n0, n1,n1,n1, n2,n2,n2,n2,n2};
    #pragma unroll
    for (int v=0;v<8;v++){
        const int e   = e0 + erow0 + v;
        const int dst = edges[2*e + 1];
        float* mp = msg + (((size_t)b*NN + dst)*CC + k)*9;
        #pragma unroll
        for (int i=0;i<9;i++) atomicAdd(&mp[i], accR[v][i]*nrm[i]);
    }
}

// ---------------- per-node product block + lin mix + skip ----------------
__global__ void __launch_bounds__(64) k_prod(
    const float* __restrict__ A_in, const float* __restrict__ h_in,
    const float* __restrict__ cg,
    const float* __restrict__ w1, const float* __restrict__ w2,
    const float* __restrict__ w3, const float* __restrict__ lin,
    float* __restrict__ h_out, float* __restrict__ out_final, int is_last)
{
    __shared__ float cgS[11*125];
    __shared__ float accS[CC*9];
    const int node = blockIdx.x;          // b*N + n
    const int c = threadIdx.x;            // channel (also output channel o)
    for (int x=c; x<11*125; x+=64) cgS[x] = cg[x];
    __syncthreads();

    const float* Ap = A_in + ((size_t)node*CC + c)*9;
    float A[9];
    #pragma unroll
    for (int i=0;i<9;i++) A[i] = Ap[i];

    float acc[9];
    #pragma unroll
    for (int l=0;l<3;l++){
        float wl = w1[l*CC + c];
        #pragma unroll
        for (int i=0;i<DIMSc[l];i++) acc[OFFc[l]+i] = wl * A[OFFc[l]+i];
    }

    float T[11][5];
    #pragma unroll
    for (int p=0;p<11;p++){
        const int l1 = PL1c[p], l2 = PL2c[p], l3 = PL3c[p];
        const int d1 = DIMSc[l1], d2 = DIMSc[l2], d3 = DIMSc[l3];
        const int o1 = OFFc[l1], o2 = OFFc[l2], o3 = OFFc[l3];
        const float wp = w2[p*CC + c];
        #pragma unroll
        for (int k=0;k<d3;k++){
            float s = 0.f;
            #pragma unroll
            for (int i=0;i<d1;i++)
                #pragma unroll
                for (int j=0;j<d2;j++)
                    s += A[o1+i]*A[o2+j]*cgS[p*125 + (i*5+j)*5 + k];
            T[p][k] = s;
            acc[o3+k] += wp * s;
        }
    }

    #pragma unroll
    for (int q=0;q<41;q++){
        const int p1 = P3Ac[q], p2 = P3Bc[q];
        const int l1 = PL1c[p2], l2 = PL2c[p2], l3 = PL3c[p2];
        const int d1 = DIMSc[l1], d2 = DIMSc[l2], d3 = DIMSc[l3];
        const int o2 = OFFc[l2], o3 = OFFc[l3];
        const float wq = w3[q*CC + c];
        #pragma unroll
        for (int k=0;k<d3;k++){
            float s = 0.f;
            #pragma unroll
            for (int i=0;i<d1;i++)
                #pragma unroll
                for (int j=0;j<d2;j++)
                    s += T[p1][i]*A[o2+j]*cgS[p2*125 + (i*5+j)*5 + k];
            acc[o3+k] += wq * s;
        }
    }

    #pragma unroll
    for (int i=0;i<9;i++) accS[c*9 + i] = acc[i];
    __syncthreads();

    const float invs = 0.125f;            // 1/sqrt(C)
    const float* hp = h_in + ((size_t)node*CC + c)*9;
    float* op = h_out + ((size_t)node*CC + c)*9;
    #pragma unroll
    for (int i=0;i<9;i++){
        const int l = (i==0) ? 0 : ((i<4) ? 1 : 2);
        const float* lp = lin + l*4096 + c;     // lin[l][cc][o=c]
        float s = 0.f;
        for (int cc=0; cc<CC; cc++) s += accS[cc*9 + i] * lp[cc*64];
        float val = s*invs + hp[i];
        op[i] = val;
        if (is_last && i==0) out_final[(size_t)node*CC + c] = val;
    }
}

// ---------------- host orchestration ----------------
extern "C" void kernel_launch(void* const* d_in, const int* in_sizes, int n_in,
                              void* d_out, int out_size, void* d_ws, size_t ws_size,
                              hipStream_t stream) {
    const int*   at    = (const int*)  d_in[0];
    const int*   edges = (const int*)  d_in[1];
    const float* pos   = (const float*)d_in[2];
    const float* emb   = (const float*)d_in[3];

    // Detect params flattening order (insertion vs sorted keys)
    const bool sorted_keys = (n_in > 4 && in_sizes[4] == 64); // fc1_b first => sorted

    char* ws = (char*)d_ws;
    size_t off = 0;
    auto take = [&](size_t bytes)->void*{
        void* p = (void*)(ws + off);
        off += (bytes + 255) & ~(size_t)255;
        return p;
    };
    float* cg  = (float*)take(11*125*sizeof(float));
    float* Y   = (float*)take((size_t)BE*9*sizeof(float));
    float* ef  = (float*)take((size_t)BE*8*sizeof(float));
    unsigned short* hid = (unsigned short*)take((size_t)BE*CC*sizeof(unsigned short));
    unsigned short* wf  = (unsigned short*)take((size_t)45056*CC*sizeof(unsigned short));
    float* tb  = (float*)take((size_t)BE*CC*TCOLS*sizeof(float));
    float* msg = (float*)take((size_t)BB*NN*CC*9*sizeof(float));
    float* hA  = (float*)take((size_t)BB*NN*CC*9*sizeof(float));
    float* hB  = (float*)take((size_t)BB*NN*CC*9*sizeof(float));

    k_cg<<<11, 128, 0, stream>>>(cg);
    k_geom<<<(BE+255)/256, 256, 0, stream>>>(pos, edges, Y, ef);
    k_init_h<<<(BB*NN*CC)/256, 256, 0, stream>>>(at, emb, hA);

    float* hin = hA; float* hout = hB;
    for (int L = 0; L < 3; L++){
        const int base = 4 + 8*L;
        const float *fc1w, *fc1b, *fc2w, *fc2b, *w1, *w2, *w3, *lin;
        if (!sorted_keys){
            fc1w=(const float*)d_in[base+0]; fc1b=(const float*)d_in[base+1];
            fc2w=(const float*)d_in[base+2]; fc2b=(const float*)d_in[base+3];
            w1  =(const float*)d_in[base+4]; w2  =(const float*)d_in[base+5];
            w3  =(const float*)d_in[base+6]; lin =(const float*)d_in[base+7];
        } else {
            fc1b=(const float*)d_in[base+0]; fc1w=(const float*)d_in[base+1];
            fc2b=(const float*)d_in[base+2]; fc2w=(const float*)d_in[base+3];
            lin =(const float*)d_in[base+4]; w1  =(const float*)d_in[base+5];
            w2  =(const float*)d_in[base+6]; w3  =(const float*)d_in[base+7];
        }
        const int npaths = (L == 0) ? 3 : 11;
        const int numel  = npaths * 4096;

        k_wconv<<<1024, 256, 0, stream>>>(fc2w, wf, numel);
        k_hid<<<(BE*CC)/256, 256, 0, stream>>>(ef, fc1w, fc1b, hid);
        k_t<<<(BE*CC)/256, 256, 0, stream>>>(hin, Y, edges, cg, tb, npaths);
        k_zero<<<(BB*NN*CC*9 + 255)/256, 256, 0, stream>>>(msg, BB*NN*CC*9);

        float n0, n1, n2;
        if (L == 0){ n0 = n1 = n2 = 0.125f; }
        else { n0 = 1.0f/sqrtf(64.f*3.f); n1 = 1.0f/sqrtf(64.f*4.f); n2 = n1; }

        k_conv<<<BE/ET, 256, 0, stream>>>(wf, fc2b, hid, tb, edges, msg, npaths, n0, n1, n2);
        k_prod<<<BB*NN, 64, 0, stream>>>(msg, hin, cg, w1, w2, w3, lin,
                                         hout, (float*)d_out, (L == 2) ? 1 : 0);
        float* tmp = hin; hin = hout; hout = tmp;
    }
}
